// BigramLanguageModel_27324581937166
// MI455X (gfx1250) — matrix operand order repaired and checked by reading, not compiled, not run on hardware
//
#include <hip/hip_runtime.h>
#include <hip/hip_bf16.h>
#include <hip/hip_fp16.h>

// ---------------------------------------------------------------------------
// Mini-GPT forward for MI455X (gfx1250, wave32, WMMA).
// All GEMMs run as v_wmma_f32_16x16x32_f16 (f16 inputs, f32 accumulate).
// Big GEMMs use 64x64 register blocking per wave: 16 WMMAs per k-step on
// 4 A-fragments x 4 B-fragments (16 WMMAs : 16 b128 loads, ~32 FLOP/B of L2).
// Weights are converted once per launch into f16 B^T ([N][K] row-major) so
// every WMMA operand fragment is a pair of contiguous 16B loads.
// Attention is fused per 16-row query tile (one wave / block), softmax in f32.
// Workspace requirement: ~160 MB.
// ---------------------------------------------------------------------------

#define DEV __device__ __forceinline__

typedef __attribute__((ext_vector_type(16))) _Float16 v16h;
typedef __attribute__((ext_vector_type(8)))  _Float16 v8h;
typedef __attribute__((ext_vector_type(8)))  float    v8f;

constexpr int B_  = 64;
constexpr int T_  = 256;
constexpr int V_  = 65;
constexpr int D_  = 384;
constexpr int H_  = 6;
constexpr int L_  = 6;
constexpr int HD_ = D_ / H_;   // 64
constexpr int FF_ = 4 * D_;    // 1536
constexpr int BT_ = B_ * T_;   // 16384
constexpr int VP_ = 80;        // V padded to a multiple of 16

// --------------------------- WMMA fragment helpers -------------------------
// A (16x32 f16, row-major source, ld = row stride in elements):
//   lane 0-15:  M=lane,    halves 0-7 -> K=k0+0..7,   halves 8-15 -> K=k0+16..23
//   lane 16-31: M=lane-16, halves 0-7 -> K=k0+8..15,  halves 8-15 -> K=k0+24..31
DEV v16h load_a16(const _Float16* A, int lda, int m, int k0, int lane) {
  const _Float16* p = A + (size_t)m * lda + k0 + ((lane & 16) ? 8 : 0);
  v8h x0 = *(const v8h*)p;
  v8h x1 = *(const v8h*)(p + 16);
  v16h r;
#pragma unroll
  for (int i = 0; i < 8; ++i) { r[i] = x0[i]; r[i + 8] = x1[i]; }
  return r;
}

// B (32x16 f16) from B^T stored row-major [N][K] (ldb = K stride):
//   lane 0-15:  N=lane,    halves 0-7 -> K=k0+0..7,   halves 8-15 -> K=k0+16..23
//   lane 16-31: N=lane-16, halves 0-7 -> K=k0+8..15,  halves 8-15 -> K=k0+24..31
DEV v16h load_b16(const _Float16* Bt, int ldb, int n, int k0, int lane) {
  const _Float16* p = Bt + (size_t)n * ldb + k0 + ((lane & 16) ? 8 : 0);
  v8h x0 = *(const v8h*)p;
  v8h x1 = *(const v8h*)(p + 16);
  v16h r;
#pragma unroll
  for (int i = 0; i < 8; ++i) { r[i] = x0[i]; r[i + 8] = x1[i]; }
  return r;
}

DEV v8f wmma_f16(v16h a, v16h b, v8f c) {
  // (neg_a, A, neg_b, B, c_mod, C, reuse_a, reuse_b)
  return __builtin_amdgcn_wmma_f32_16x16x32_f16(false, a, false, b,
                                                (short)0, c, false, false);
}

DEV v8f zero8f() {
  v8f z;
#pragma unroll
  for (int i = 0; i < 8; ++i) z[i] = 0.0f;
  return z;
}

// ---- 16x16 core (head GEMM only) ----
DEV v8f gemm_core16(const _Float16* __restrict__ A, int lda,
                    const _Float16* __restrict__ Bt, int ldb,
                    int m0, int n0, int K, int lane) {
  v8f acc = zero8f();
  const int m = m0 + (lane & 15);
  const int n = n0 + (lane & 15);
  for (int k0 = 0; k0 < K; k0 += 32) {
    if (k0 + 32 < K) {
      __builtin_prefetch(A  + (size_t)m * lda + k0 + 32, 0, 1);
      __builtin_prefetch(Bt + (size_t)n * ldb + k0 + 32, 0, 1);
    }
    v16h a = load_a16(A, lda, m, k0, lane);
    v16h b = load_b16(Bt, ldb, n, k0, lane);
    acc = wmma_f16(a, b, acc);
  }
  return acc;
}

// ---- 64x64 register-blocked core (all big GEMMs) ----
// C layout per subtile (i,j): element r -> row m0+i*16 + r + (lane&16?8:0),
//                                         col n0+j*16 + (lane&15).
struct Acc44 { v8f t[4][4]; };

DEV void gemm_core64(const _Float16* __restrict__ A, int lda,
                     const _Float16* __restrict__ Bt, int ldb,
                     int m0, int n0, int K, int lane, Acc44& acc) {
#pragma unroll
  for (int i = 0; i < 4; ++i)
#pragma unroll
    for (int j = 0; j < 4; ++j) acc.t[i][j] = zero8f();

  const int ml = lane & 15;
  for (int k0 = 0; k0 < K; k0 += 32) {
    if (k0 + 64 < K) {   // uniform; lowers to global_prefetch_b8
      __builtin_prefetch(A  + (size_t)(m0 + ml) * lda + k0 + 64, 0, 1);
      __builtin_prefetch(Bt + (size_t)(n0 + ml) * ldb + k0 + 64, 0, 1);
    }
    v16h a[4], b[4];
#pragma unroll
    for (int i = 0; i < 4; ++i)
      a[i] = load_a16(A, lda, m0 + i * 16 + ml, k0, lane);
#pragma unroll
    for (int j = 0; j < 4; ++j)
      b[j] = load_b16(Bt, ldb, n0 + j * 16 + ml, k0, lane);
#pragma unroll
    for (int i = 0; i < 4; ++i)
#pragma unroll
      for (int j = 0; j < 4; ++j)
        acc.t[i][j] = wmma_f16(a[i], b[j], acc.t[i][j]);
  }
}

// ------------------------------ weight prep --------------------------------
// wq/wk/wv [L][H][D][HD] f32  ->  WqkvT [L][3D][D] f16 (B^T: row n = out col)
__global__ void cvt_qkv_kernel(const float* __restrict__ wq,
                               const float* __restrict__ wk,
                               const float* __restrict__ wv,
                               _Float16* __restrict__ dst) {
  size_t i = (size_t)blockIdx.x * blockDim.x + threadIdx.x;
  size_t total = (size_t)L_ * 3 * D_ * D_;
  if (i >= total) return;
  int dcol = (int)(i % D_);            // K index (input dim d)
  size_t rest = i / D_;
  int c = (int)(rest % (3 * D_));      // out col: which*D + head*HD + e
  int l = (int)(rest / (3 * D_));
  int which = c / D_;
  int w = c % D_;
  int head = w / HD_, e = w % HD_;
  const float* src = (which == 0) ? wq : (which == 1) ? wk : wv;
  float v = src[(((size_t)l * H_ + head) * D_ + dcol) * HD_ + e];
  dst[i] = (_Float16)v;
}

// src [R][C] f32 (row-major) -> dst [Cp][R] f16 (B^T), zero pad for c >= C.
__global__ void cvt_transpose_kernel(const float* __restrict__ src,
                                     _Float16* __restrict__ dst,
                                     int R, int C, int Cp) {
  size_t i = (size_t)blockIdx.x * blockDim.x + threadIdx.x;
  size_t total = (size_t)Cp * R;
  if (i >= total) return;
  int r = (int)(i % R);
  int c = (int)(i / R);
  dst[i] = (_Float16)((c < C) ? src[(size_t)r * C + c] : 0.0f);
}

// ------------------------------- embedding ---------------------------------
__global__ void embed_kernel(const int* __restrict__ idx,
                             const float* __restrict__ tok,
                             const float* __restrict__ pos,
                             float* __restrict__ x) {
  size_t i = (size_t)blockIdx.x * blockDim.x + threadIdx.x;
  if (i >= (size_t)BT_ * D_) return;
  int d = (int)(i % D_);
  int m = (int)(i / D_);
  int t = m % T_;
  int token = idx[m];
  x[i] = tok[(size_t)token * D_ + d] + pos[(size_t)t * D_ + d];
}

// ------------------------------- layernorm ---------------------------------
// One row per block (128 threads, D = 3*128). f32 stats, f16 output.
__global__ __launch_bounds__(128) void layernorm_kernel(
    const float* __restrict__ x, const float* __restrict__ w,
    const float* __restrict__ bias, _Float16* __restrict__ out) {
  __shared__ float sh_s[128];
  __shared__ float sh_q[128];
  const int tid = threadIdx.x;
  const int row = blockIdx.x;
  const float* xr = x + (size_t)row * D_;
  float s = 0.f, q = 0.f;
#pragma unroll
  for (int i = 0; i < 3; ++i) {
    float v = xr[tid + i * 128];
    s += v; q += v * v;
  }
  sh_s[tid] = s; sh_q[tid] = q;
  __syncthreads();
  for (int st = 64; st > 0; st >>= 1) {
    if (tid < st) { sh_s[tid] += sh_s[tid + st]; sh_q[tid] += sh_q[tid + st]; }
    __syncthreads();
  }
  float mean = sh_s[0] * (1.0f / D_);
  float var  = sh_q[0] * (1.0f / D_) - mean * mean;
  float rstd = rsqrtf(var + 1e-5f);
#pragma unroll
  for (int i = 0; i < 3; ++i) {
    int d = tid + i * 128;
    out[(size_t)row * D_ + d] =
        (_Float16)((xr[d] - mean) * rstd * w[d] + bias[d]);
  }
}

// ------------------------------- GEMM kernels ------------------------------
// QKV: A = ln1(x) f16 [BT,D]; Bt = WqkvT(layer) [3D][D]; 64x64 macro-tiles.
// The 64-wide N band never straddles q/k/v or head boundaries (D, HD = 64k).
// Epilogue scatters f16 into q [B,H,T,HD], k [B,H,T,HD], vT [B,H,HD,T].
__global__ __launch_bounds__(32) void gemm_qkv_kernel(
    const _Float16* __restrict__ A, const _Float16* __restrict__ WqkvT,
    _Float16* __restrict__ q, _Float16* __restrict__ k,
    _Float16* __restrict__ vT) {
  const int ntiles = 3 * D_ / 64;   // 18
  const int lane = threadIdx.x;
  const int nt = blockIdx.x % ntiles;
  const int mt = blockIdx.x / ntiles;
  const int m0 = mt * 64, n0 = nt * 64;
  Acc44 acc;
  gemm_core64(A, D_, WqkvT, D_, m0, n0, D_, lane, acc);

  const int which = n0 / D_;              // uniform per block
  const int head  = (n0 % D_) / HD_;      // uniform per block
  const int bb    = m0 / T_;              // 64-row tile stays in one sequence
  const int t0    = m0 % T_;
  const int hi8   = (lane & 16) ? 8 : 0;
  const int nl    = lane & 15;
  const size_t bh = (size_t)bb * H_ + head;
#pragma unroll
  for (int j = 0; j < 4; ++j) {
    const int e = j * 16 + nl;            // 0..63 within the head
#pragma unroll
    for (int i = 0; i < 4; ++i) {
#pragma unroll
      for (int r = 0; r < 8; ++r) {
        const int t = t0 + i * 16 + hi8 + r;
        const _Float16 val = (_Float16)acc.t[i][j][r];
        if (which == 0)      q[(bh * T_ + t) * HD_ + e] = val;
        else if (which == 1) k[(bh * T_ + t) * HD_ + e] = val;
        else                 vT[(bh * HD_ + e) * T_ + t] = val;
      }
    }
  }
}

// xio[m,n] += A*Bt + bias[n]   (proj and FF2 residual adds, f32 in-place)
__global__ __launch_bounds__(32) void gemm_bias_res_f32_kernel(
    const _Float16* __restrict__ A, int lda,
    const _Float16* __restrict__ Bt, int ldb,
    const float* __restrict__ bias,
    float* __restrict__ xio, int ldo, int K, int ntiles64) {
  const int lane = threadIdx.x;
  const int nt = blockIdx.x % ntiles64;
  const int mt = blockIdx.x / ntiles64;
  const int m0 = mt * 64, n0 = nt * 64;
  Acc44 acc;
  gemm_core64(A, lda, Bt, ldb, m0, n0, K, lane, acc);
  const int hi8 = (lane & 16) ? 8 : 0;
  const int nl  = lane & 15;
#pragma unroll
  for (int j = 0; j < 4; ++j) {
    const int n = n0 + j * 16 + nl;
    const float bn = bias[n];
#pragma unroll
    for (int i = 0; i < 4; ++i) {
      const int rb = m0 + i * 16 + hi8;
#pragma unroll
      for (int r = 0; r < 8; ++r) {
        size_t o = (size_t)(rb + r) * ldo + n;
        xio[o] += acc.t[i][j][r] + bn;
      }
    }
  }
}

// out[m,n] = relu(A*Bt + bias[n]) as f16   (FF1)
__global__ __launch_bounds__(32) void gemm_bias_relu_f16_kernel(
    const _Float16* __restrict__ A, int lda,
    const _Float16* __restrict__ Bt, int ldb,
    const float* __restrict__ bias,
    _Float16* __restrict__ out, int ldo, int K, int ntiles64) {
  const int lane = threadIdx.x;
  const int nt = blockIdx.x % ntiles64;
  const int mt = blockIdx.x / ntiles64;
  const int m0 = mt * 64, n0 = nt * 64;
  Acc44 acc;
  gemm_core64(A, lda, Bt, ldb, m0, n0, K, lane, acc);
  const int hi8 = (lane & 16) ? 8 : 0;
  const int nl  = lane & 15;
#pragma unroll
  for (int j = 0; j < 4; ++j) {
    const int n = n0 + j * 16 + nl;
    const float bn = bias[n];
#pragma unroll
    for (int i = 0; i < 4; ++i) {
      const int rb = m0 + i * 16 + hi8;
#pragma unroll
      for (int r = 0; r < 8; ++r)
        out[(size_t)(rb + r) * ldo + n] =
            (_Float16)fmaxf(acc.t[i][j][r] + bn, 0.0f);
    }
  }
}

// logits[m, n<V] = A*Bt + bias[n]  (head; Bt padded to VP rows; 16-wide tiles)
__global__ __launch_bounds__(32) void gemm_head_kernel(
    const _Float16* __restrict__ A, const _Float16* __restrict__ Bt,
    const float* __restrict__ bias, float* __restrict__ out) {
  const int ntiles = VP_ / 16;   // 5
  const int lane = threadIdx.x;
  const int nt = blockIdx.x % ntiles;
  const int mt = blockIdx.x / ntiles;
  v8f acc = gemm_core16(A, D_, Bt, D_, mt * 16, nt * 16, D_, lane);
  const int n = nt * 16 + (lane & 15);
  const int rbase = mt * 16 + ((lane & 16) ? 8 : 0);
  if (n < V_) {
    const float bn = bias[n];
#pragma unroll
    for (int r = 0; r < 8; ++r)
      out[(size_t)(rbase + r) * V_ + n] = acc[r] + bn;
  }
}

// ------------------------------- attention ---------------------------------
// One wave per (b, h, 16-row query tile). S = Q*K^T (WMMA), causal softmax in
// f32 registers, P staged through LDS (C-layout -> A-fragment transpose),
// O = P*V (WMMA).  scale = D^-0.5 per the reference.
__global__ __launch_bounds__(32) void attn_kernel(
    const _Float16* __restrict__ q, const _Float16* __restrict__ k,
    const _Float16* __restrict__ vT, _Float16* __restrict__ o) {
  __shared__ __align__(16) _Float16 Pl[16 * T_];   // 8 KB
  const int lane = threadIdx.x;
  const int qt = blockIdx.x & 15;          // T/16 = 16 query tiles
  const int bh = blockIdx.x >> 4;
  const int bb = bh / H_, head = bh % H_;
  const _Float16* Q  = q  + (size_t)bh * T_ * HD_;   // [t][e]
  const _Float16* Kt = k  + (size_t)bh * T_ * HD_;   // B^T [s][e]
  const _Float16* Vt = vT + (size_t)bh * HD_ * T_;   // B^T [e][s]
  const int m0 = qt * 16;
  const int mrow = m0 + (lane & 15);
  const int hi8 = (lane & 16) ? 8 : 0;

  // zero LDS (covers the unused half-kstep tile when qt is even)
  v8h zh;
#pragma unroll
  for (int i = 0; i < 8; ++i) zh[i] = (_Float16)0.0f;
#pragma unroll
  for (int i = 0; i < 16; ++i) ((v8h*)Pl)[lane + i * 32] = zh;
  __syncthreads();

  // Q fragments (A: 16 x 64 -> 2 k-steps)
  v16h qa0 = load_a16(Q, HD_, mrow, 0, lane);
  v16h qa1 = load_a16(Q, HD_, mrow, 32, lane);

  const float scale = rsqrtf((float)D_);   // n_embd**-0.5 (matches reference)
  float rowmax[8], rowsum[8];
#pragma unroll
  for (int r = 0; r < 8; ++r) { rowmax[r] = -1e30f; rowsum[r] = 0.0f; }

  // Pass 1: scores for key tiles st <= qt (uniform branch -> EXEC all ones)
  v8f S[16];
#pragma unroll
  for (int st = 0; st < 16; ++st) {
    S[st] = zero8f();
    if (st <= qt) {
      int ncol = st * 16 + (lane & 15);
      v16h b0 = load_b16(Kt, HD_, ncol, 0, lane);
      v16h b1 = load_b16(Kt, HD_, ncol, 32, lane);
      v8f acc = zero8f();
      acc = wmma_f16(qa0, b0, acc);
      acc = wmma_f16(qa1, b1, acc);
#pragma unroll
      for (int r = 0; r < 8; ++r) {
        float v = acc[r] * scale;
        if (st == qt && (lane & 15) > (r + hi8)) v = -1e30f;  // causal mask
        acc[r] = v;
        rowmax[r] = fmaxf(rowmax[r], v);
      }
      S[st] = acc;
    }
  }

  // row max across the 16-lane group holding each row
#pragma unroll
  for (int r = 0; r < 8; ++r) {
    float v = rowmax[r];
#pragma unroll
    for (int off = 1; off < 16; off <<= 1) v = fmaxf(v, __shfl_xor(v, off, 32));
    rowmax[r] = v;
  }

  // exp + row sum
#pragma unroll
  for (int st = 0; st < 16; ++st) {
    if (st <= qt) {
#pragma unroll
      for (int r = 0; r < 8; ++r) {
        float p = __expf(S[st][r] - rowmax[r]);
        S[st][r] = p;
        rowsum[r] += p;
      }
    }
  }
#pragma unroll
  for (int r = 0; r < 8; ++r) {
    float v = rowsum[r];
#pragma unroll
    for (int off = 1; off < 16; off <<= 1) v += __shfl_xor(v, off, 32);
    rowsum[r] = v;
  }
  float rinv[8];
#pragma unroll
  for (int r = 0; r < 8; ++r) rinv[r] = 1.0f / rowsum[r];  // >= 1 (diag term)

  // write normalized P (f16, row-major [16][T]) to LDS
#pragma unroll
  for (int st = 0; st < 16; ++st) {
    if (st <= qt) {
      int col = st * 16 + (lane & 15);
#pragma unroll
      for (int r = 0; r < 8; ++r)
        Pl[(r + hi8) * T_ + col] = (_Float16)(S[st][r] * rinv[r]);
    }
  }
  __syncthreads();

  // O = P * V  (k-steps of 32 over the valid key range)
  v8f O[4];
#pragma unroll
  for (int nt = 0; nt < 4; ++nt) O[nt] = zero8f();
  const int nksteps = (qt + 2) >> 1;   // ceil((qt+1)/2), uniform
  for (int ks = 0; ks < nksteps; ++ks) {
    v16h pa = load_a16(Pl, T_, (lane & 15), ks * 32, lane);
#pragma unroll
    for (int nt = 0; nt < 4; ++nt) {
      int ncol = nt * 16 + (lane & 15);
      v16h vb = load_b16(Vt, T_, ncol, ks * 32, lane);
      O[nt] = wmma_f16(pa, vb, O[nt]);
    }
  }

  // store O into concatenated-heads layout o [BT, D], col = head*HD + e
#pragma unroll
  for (int nt = 0; nt < 4; ++nt) {
    int e = nt * 16 + (lane & 15);
    int colD = head * HD_ + e;
#pragma unroll
    for (int r = 0; r < 8; ++r) {
      int t = m0 + r + hi8;
      o[((size_t)bb * T_ + t) * D_ + colD] = (_Float16)O[nt][r];
    }
  }
}

// ------------------------------- host driver -------------------------------
extern "C" void kernel_launch(void* const* d_in, const int* in_sizes, int n_in,
                              void* d_out, int out_size, void* d_ws,
                              size_t ws_size, hipStream_t stream) {
  (void)in_sizes; (void)n_in; (void)out_size; (void)ws_size;
  const int*   idx     = (const int*)  d_in[0];
  const float* tok_emb = (const float*)d_in[1];
  const float* pos_emb = (const float*)d_in[2];
  const float* ln1_w   = (const float*)d_in[3];
  const float* ln1_b   = (const float*)d_in[4];
  const float* wq      = (const float*)d_in[5];
  const float* wk      = (const float*)d_in[6];
  const float* wv      = (const float*)d_in[7];
  const float* proj_w  = (const float*)d_in[8];
  const float* proj_b  = (const float*)d_in[9];
  const float* ln2_w   = (const float*)d_in[10];
  const float* ln2_b   = (const float*)d_in[11];
  const float* ff_w1   = (const float*)d_in[12];
  const float* ff_b1   = (const float*)d_in[13];
  const float* ff_w2   = (const float*)d_in[14];
  const float* ff_b2   = (const float*)d_in[15];
  const float* lnf_w   = (const float*)d_in[16];
  const float* lnf_b   = (const float*)d_in[17];
  const float* head_w  = (const float*)d_in[18];
  const float* head_b  = (const float*)d_in[19];
  float* out = (float*)d_out;

  // workspace carve (256B aligned), ~160 MB total
  char* p = (char*)d_ws;
  auto carve = [&](size_t bytes) -> char* {
    char* r = p;
    p += (bytes + 255) & ~(size_t)255;
    return r;
  };
  _Float16* WqkvT  = (_Float16*)carve((size_t)L_ * 3 * D_ * D_ * 2);
  _Float16* WprojT = (_Float16*)carve((size_t)L_ * D_ * D_ * 2);
  _Float16* Wff1T  = (_Float16*)carve((size_t)L_ * FF_ * D_ * 2);
  _Float16* Wff2T  = (_Float16*)carve((size_t)L_ * D_ * FF_ * 2);
  _Float16* WheadT = (_Float16*)carve((size_t)VP_ * D_ * 2);
  float*    x      = (float*)   carve((size_t)BT_ * D_ * 4);
  _Float16* hbuf   = (_Float16*)carve((size_t)BT_ * D_ * 2);
  _Float16* qb     = (_Float16*)carve((size_t)B_ * H_ * T_ * HD_ * 2);
  _Float16* kb     = (_Float16*)carve((size_t)B_ * H_ * T_ * HD_ * 2);
  _Float16* vtb    = (_Float16*)carve((size_t)B_ * H_ * T_ * HD_ * 2);
  _Float16* obuf   = (_Float16*)carve((size_t)BT_ * D_ * 2);
  _Float16* ffb    = (_Float16*)carve((size_t)BT_ * FF_ * 2);

  auto blocks = [](size_t total, int bs) { return (unsigned)((total + bs - 1) / bs); };

  // ---- weight conversion (f32 -> f16 B^T layouts) ----
  cvt_qkv_kernel<<<blocks((size_t)L_ * 3 * D_ * D_, 256), 256, 0, stream>>>(
      wq, wk, wv, WqkvT);
  for (int l = 0; l < L_; ++l) {
    cvt_transpose_kernel<<<blocks((size_t)D_ * D_, 256), 256, 0, stream>>>(
        proj_w + (size_t)l * D_ * D_, WprojT + (size_t)l * D_ * D_, D_, D_, D_);
    cvt_transpose_kernel<<<blocks((size_t)FF_ * D_, 256), 256, 0, stream>>>(
        ff_w1 + (size_t)l * D_ * FF_, Wff1T + (size_t)l * FF_ * D_, D_, FF_, FF_);
    cvt_transpose_kernel<<<blocks((size_t)D_ * FF_, 256), 256, 0, stream>>>(
        ff_w2 + (size_t)l * FF_ * D_, Wff2T + (size_t)l * D_ * FF_, FF_, D_, D_);
  }
  cvt_transpose_kernel<<<blocks((size_t)VP_ * D_, 256), 256, 0, stream>>>(
      head_w, WheadT, D_, V_, VP_);

  // ---- embedding ----
  embed_kernel<<<blocks((size_t)BT_ * D_, 256), 256, 0, stream>>>(
      idx, tok_emb, pos_emb, x);

  const unsigned mt64 = BT_ / 64;   // 256 macro M-tiles
  // ---- transformer layers ----
  for (int l = 0; l < L_; ++l) {
    layernorm_kernel<<<BT_, 128, 0, stream>>>(x, ln1_w + (size_t)l * D_,
                                              ln1_b + (size_t)l * D_, hbuf);
    gemm_qkv_kernel<<<mt64 * (3 * D_ / 64), 32, 0, stream>>>(
        hbuf, WqkvT + (size_t)l * 3 * D_ * D_, qb, kb, vtb);
    attn_kernel<<<B_ * H_ * (T_ / 16), 32, 0, stream>>>(qb, kb, vtb, obuf);
    gemm_bias_res_f32_kernel<<<mt64 * (D_ / 64), 32, 0, stream>>>(
        obuf, D_, WprojT + (size_t)l * D_ * D_, D_,
        proj_b + (size_t)l * D_, x, D_, D_, D_ / 64);
    layernorm_kernel<<<BT_, 128, 0, stream>>>(x, ln2_w + (size_t)l * D_,
                                              ln2_b + (size_t)l * D_, hbuf);
    gemm_bias_relu_f16_kernel<<<mt64 * (FF_ / 64), 32, 0, stream>>>(
        hbuf, D_, Wff1T + (size_t)l * FF_ * D_, D_,
        ff_b1 + (size_t)l * FF_, ffb, FF_, D_, FF_ / 64);
    gemm_bias_res_f32_kernel<<<mt64 * (D_ / 64), 32, 0, stream>>>(
        ffb, FF_, Wff2T + (size_t)l * D_ * FF_, FF_,
        ff_b2 + (size_t)l * D_, x, D_, FF_, D_ / 64);
  }

  // ---- final LN + head ----
  layernorm_kernel<<<BT_, 128, 0, stream>>>(x, lnf_w, lnf_b, hbuf);
  gemm_head_kernel<<<(BT_ / 16) * (VP_ / 16), 32, 0, stream>>>(
      hbuf, WheadT, head_b, out);
}
